// ESN_73770358276577
// MI455X (gfx1250) — matrix operand order repaired and checked by reading, not compiled, not run on hardware
//
#include <hip/hip_runtime.h>
#include <cstddef>
#include <math.h>

// Problem constants (match reference setup_inputs)
#define B_DIM 16
#define H_DIM 128     // D_IN  (K of the GEMM)
#define P_DIM 2048    // D_STATE
#define L_DIM 2048    // sequence length
#define LEAK  0.5f

typedef __attribute__((ext_vector_type(16))) _Float16 v16h;
typedef __attribute__((ext_vector_type(8)))  _Float16 v8h;
typedef __attribute__((ext_vector_type(8)))  float    v8f;

// ---------------------------------------------------------------------------
// Kernel 0: convert w_in (f32, [P,H] row-major) -> f16 copy in workspace.
// 512 KB; w_in stays L2-resident for the whole GEMM.
// ---------------------------------------------------------------------------
__global__ __launch_bounds__(256) void esn_wconv(const float* __restrict__ w,
                                                 _Float16* __restrict__ o, int n) {
  int i = blockIdx.x * blockDim.x + threadIdx.x;
  if (i < n) o[i] = (_Float16)w[i];
}

// ---------------------------------------------------------------------------
// Kernel 1: PRE[b,p,t] = sum_h w_in[p,h]*u[b,h,t] + bias[p]  via WMMA f16.
// One wave per (b, t-tile, p-group); each wave computes PT p-tiles of 16x16,
// reusing the u-tile B operand (held in 32 VGPRs) across all PT tiles.
// K = H = 128 -> 4 chained v_wmma_f32_16x16x32_f16 per tile.
// ---------------------------------------------------------------------------
#define PT 8                       // p-tiles per wave
#define TT (L_DIM / 16)            // 128 t-tiles
#define PG (P_DIM / (16 * PT))     // 16 p-groups

__global__ __launch_bounds__(32) void esn_gemm_pre(
    const float*    __restrict__ u,      // [B, H, L] f32
    const _Float16* __restrict__ wf16,   // [P, H]    f16
    const float*    __restrict__ bias,   // [P]
    float*          __restrict__ out)    // [B, P, L] f32 (PRE, pre-activation)
{
  const int wid   = blockIdx.x;
  const int b     = wid / (TT * PG);
  const int rem   = wid % (TT * PG);
  const int tt    = rem / PG;
  const int pg    = rem % PG;
  const int t0    = tt * 16;
  const int pbase = pg * (16 * PT);

  const int lane = threadIdx.x;   // wave32
  const int lo   = lane & 15;
  const int hi   = lane >> 4;

  // --- B operand: u tile (K=h rows x N=t cols), 4 chunks of K=32.
  // Layout (this code's contiguous k order, same as A): lane holds column N = lane%16;
  // lanes 0-15 hold K = chunk*32 + 0..15, lanes 16-31 hold K = chunk*32 + 16..31,
  // packed 2 f16 per VGPR (v16h element e <-> K = hi*16 + e).
  // Gather: for a fixed e, lanes 0-15 read 64 contiguous bytes (one line),
  // lanes 16-31 another line -> effectively coalesced; u is L2-resident.
  v16h Bop[4];
  const float* up = u + (size_t)b * H_DIM * L_DIM + t0 + lo;
  #pragma unroll
  for (int kk = 0; kk < 4; ++kk) {
    v16h bv;
    const int kb = kk * 32 + hi * 16;
    #pragma unroll
    for (int e = 0; e < 16; ++e)
      bv[e] = (_Float16)up[(size_t)(kb + e) * L_DIM];
    Bop[kk] = bv;
  }

  #pragma unroll
  for (int pt = 0; pt < PT; ++pt) {
    const int p0 = pbase + pt * 16;

    // C/D layout: vgpr j, lane -> row M = j + hi*8 (p), col N = lo (t).
    // Fold bias into the accumulator init.
    v8f c;
    #pragma unroll
    for (int j = 0; j < 8; ++j) c[j] = bias[p0 + hi * 8 + j];

    // A operand: w_in tile (M=p rows x K=h). Layout (this code's contiguous k order, same as B):
    // lane holds row M = lane%16; per K=32 chunk, elements are one contiguous
    // 16-f16 run: K = kb..kb+7 (v0..v3) and K = kb+8..kb+15 (v4..v7),
    // kb = (lane/16)*16 -> two 16-byte loads per chunk.
    const _Float16* arow = wf16 + (size_t)(p0 + lo) * H_DIM;
    #pragma unroll
    for (int kk = 0; kk < 4; ++kk) {
      const int kb = kk * 32 + hi * 16;
      v8h alo = *(const v8h*)(arow + kb);
      v8h ahi = *(const v8h*)(arow + kb + 8);
      v16h av;
      #pragma unroll
      for (int e = 0; e < 8; ++e) { av[e] = alo[e]; av[8 + e] = ahi[e]; }
      // (neg_a, A, neg_b, B, c_mod, C, reuse_a, reuse_b)
      c = __builtin_amdgcn_wmma_f32_16x16x32_f16(false, av, false, Bop[kk],
                                                 (short)0, c, false, false);
    }

    // Store: for fixed j, lanes 0-15 write 64 contiguous bytes of one p-row,
    // lanes 16-31 the row p+8 -> fully coalesced.
    float* orow = out + (size_t)(b * P_DIM + p0 + hi * 8) * L_DIM + t0 + lo;
    #pragma unroll
    for (int j = 0; j < 8; ++j) orow[(size_t)j * L_DIM] = c[j];
  }
}

// ---------------------------------------------------------------------------
// Kernel 2: in-place leaky-tanh scan over t.
//   x_t = (1-LEAK)*x_{t-1} + LEAK*tanh(PRE[b,p,t] + w_diag[p]*x_{t-1})
// Each block owns 64 consecutive p of one b. Tiles of 16 t are staged through
// LDS so global traffic is float4-coalesced; each thread then privately scans
// its own p-row (stride-17 padding -> bank-conflict free).
// ---------------------------------------------------------------------------
#define RBLK 64

__global__ __launch_bounds__(RBLK) void esn_recurrence(
    const float* __restrict__ w_hh,   // [P,P], only diagonal used
    const float* __restrict__ x0,     // [P]
    float*       __restrict__ out)    // [B, P, L] in/out (PRE -> x)
{
  __shared__ float tile[RBLK][17];

  const int bb  = blockIdx.x;
  const int b   = bb / (P_DIM / RBLK);
  const int pg  = bb % (P_DIM / RBLK);
  const int p0  = pg * RBLK;
  const int tid = threadIdx.x;
  const int p   = p0 + tid;

  const float wd = w_hh[(size_t)p * P_DIM + p];  // diag(w_hh)
  float x = x0[p];

  float* base = out + (size_t)(b * P_DIM + p0) * L_DIM;
  const int rrow = tid >> 2;          // 0..15
  const int tq   = (tid & 3) << 2;    // 0,4,8,12

  for (int tc = 0; tc < L_DIM / 16; ++tc) {
    // coalesced load: 64 rows x 16 t, float4 per thread x 4 passes
    #pragma unroll
    for (int r = 0; r < 4; ++r) {
      const int row = r * 16 + rrow;
      const float4 v = *reinterpret_cast<const float4*>(
          base + (size_t)row * L_DIM + tc * 16 + tq);
      tile[row][tq + 0] = v.x; tile[row][tq + 1] = v.y;
      tile[row][tq + 2] = v.z; tile[row][tq + 3] = v.w;
    }
    __syncthreads();

    // private sequential scan of this thread's p-row
    #pragma unroll
    for (int i = 0; i < 16; ++i) {
      const float pre = tile[tid][i] + wd * x;   // PRE already includes bias
      x = (1.0f - LEAK) * x + LEAK * tanhf(pre);
      tile[tid][i] = x;
    }
    __syncthreads();

    // coalesced store back
    #pragma unroll
    for (int r = 0; r < 4; ++r) {
      const int row = r * 16 + rrow;
      float4 v;
      v.x = tile[row][tq + 0]; v.y = tile[row][tq + 1];
      v.z = tile[row][tq + 2]; v.w = tile[row][tq + 3];
      *reinterpret_cast<float4*>(base + (size_t)row * L_DIM + tc * 16 + tq) = v;
    }
    __syncthreads();
  }
}

// ---------------------------------------------------------------------------
extern "C" void kernel_launch(void* const* d_in, const int* in_sizes, int n_in,
                              void* d_out, int out_size, void* d_ws, size_t ws_size,
                              hipStream_t stream) {
  const float* u    = (const float*)d_in[0];  // [B, D_IN, L]
  const float* w_in = (const float*)d_in[1];  // [P, D_IN]
  const float* w_hh = (const float*)d_in[2];  // [P, P]
  const float* bias = (const float*)d_in[3];  // [P]
  const float* x0   = (const float*)d_in[4];  // [P]
  float* out = (float*)d_out;                 // [B, P, L]

  _Float16* wf16 = (_Float16*)d_ws;           // 512 KB scratch

  const int n = P_DIM * H_DIM;
  esn_wconv<<<(n + 255) / 256, 256, 0, stream>>>(w_in, wf16, n);

  esn_gemm_pre<<<B_DIM * TT * PG, 32, 0, stream>>>(u, wf16, bias, out);

  esn_recurrence<<<B_DIM * (P_DIM / RBLK), RBLK, 0, stream>>>(w_hh, x0, out);
}
